// CompositionalCodebookLayer2_29772713295964
// MI455X (gfx1250) — matrix operand order repaired and checked by reading, not compiled, not run on hardware
//
#include <hip/hip_runtime.h>
#include <hip/hip_bf16.h>

// Product-quantization layer for MI455X (gfx1250, wave32, WMMA).
// x: (4,2048,1024) f32; codebook: (16,2048,64) f32; out: (4,2048,1024) f32.
// Per chunk c: dots = (8192x64) x (64x2048) via v_wmma_f32_16x16x32_bf16,
// argmin_n(cbsq[n] - 2*dot) (xsq dropped: constant per row), gather codeword.
// N-sweep processes 2x16 columns per body and is unrolled x2 again by pragma
// (4 tiles / 64 columns per hardware-loop iteration) so the bf16 WMMA hazard
// slots and B-load latency are covered by the neighboring tile's work.

typedef __attribute__((ext_vector_type(16))) __bf16 v16bf;
typedef __attribute__((ext_vector_type(8)))  float  v8f;

#define PQ_C    16
#define PQ_N    2048
#define PQ_SUB  64
#define PQ_D    1024
#define PQ_M    8192        // B*S
#define ROWS_PER_WG 128     // 8 waves * 16 rows

// ---------------------------------------------------------------------------
// Prep: codebook f32 -> bf16 (for WMMA B operands) + per-codeword squared norm.
// One thread per (c,n) codeword row of 64 floats.
// ---------------------------------------------------------------------------
__global__ void pq_prep_kernel(const float* __restrict__ cb,
                               __bf16* __restrict__ cb16,
                               float* __restrict__ cbsq) {
    int rid = blockIdx.x * blockDim.x + threadIdx.x;     // 0 .. 16*2048-1
    if (rid >= PQ_C * PQ_N) return;
    const float4* src = reinterpret_cast<const float4*>(cb + (size_t)rid * PQ_SUB);
    __bf16* dst = cb16 + (size_t)rid * PQ_SUB;
    float s = 0.f;
#pragma unroll
    for (int q = 0; q < 16; ++q) {
        float4 v = src[q];
        s += v.x * v.x + v.y * v.y + v.z * v.z + v.w * v.w;
        dst[q * 4 + 0] = (__bf16)v.x;
        dst[q * 4 + 1] = (__bf16)v.y;
        dst[q * 4 + 2] = (__bf16)v.z;
        dst[q * 4 + 3] = (__bf16)v.w;
    }
    cbsq[rid] = s;
}

// ---------------------------------------------------------------------------
// Main kernel: grid.x = (PQ_M/ROWS_PER_WG) * PQ_C = 64 * 16.
// Each wave32 owns a 16-row tile; sweeps N in 16-column WMMA tiles.
// ---------------------------------------------------------------------------
__global__ __launch_bounds__(256) void pq_main_kernel(
        const float*  __restrict__ x,
        const float*  __restrict__ cb,      // f32 codebook for final gather
        const __bf16* __restrict__ cb16,    // bf16 codebook for WMMA
        const float*  __restrict__ cbsq,
        float* __restrict__ out) {

    const int wg    = blockIdx.x;
    const int c     = wg >> 6;                  // chunk 0..15
    const int mtile = wg & 63;                  // row tile 0..63
    const int wave  = threadIdx.x >> 5;         // 0..7 (wave32)
    const int lane  = threadIdx.x & 31;
    const int hv    = lane >> 4;                // half-wave selector
    const int lrow  = lane & 15;
    const int rowBase = mtile * ROWS_PER_WG + wave * 16;

    // ---- Build A fragments once (rows rowBase..+15, K=0..63 of chunk c) ----
    // A 16x32 layout (matches B): lanes 0-15 row M=lane; elems 0..15 = K 0..15,
    // lanes 16-31 hold K 16..31.
    const float* xrow = x + (size_t)(rowBase + lrow) * PQ_D + c * PQ_SUB;
    v16bf a0, a1;
    const int kc = hv * 16;
#pragma unroll
    for (int i = 0; i < 8; ++i) {
        a0[i]     = (__bf16)xrow[kc +  0 + i];
        a0[i + 8] = (__bf16)xrow[kc +  8 + i];
        a1[i]     = (__bf16)xrow[kc + 32 + i];
        a1[i + 8] = (__bf16)xrow[kc + 40 + i];
    }

    const __bf16* cbc   = cb16 + (size_t)c * PQ_N * PQ_SUB;
    const float*  cbsqc = cbsq + c * PQ_N;

    float best[8];
    int   bidx[8];
#pragma unroll
    for (int r = 0; r < 8; ++r) { best[r] = 3.4e38f; bidx[r] = 0; }

    // ---- Sweep all N=2048 codewords; 2x16 columns per body, unrolled x2 ----
#pragma unroll 2
    for (int n0 = 0; n0 < PQ_N; n0 += 32) {
        // B 32x16 bf16 layout: lane column N = lane&15; lanes 0-15 hold
        // K 0..15, lanes 16-31 hold K 16..31 (16 contiguous bf16 = 2x b128).
        const __bf16* brow0 = cbc + (size_t)(n0 + lrow) * PQ_SUB;
        const __bf16* brow1 = brow0 + 16 * PQ_SUB;
        __builtin_prefetch((const void*)(brow0 + 32 * PQ_SUB), 0, 3);
        __builtin_prefetch((const void*)(brow1 + 32 * PQ_SUB), 0, 3);

        union { uint4 u[2]; v16bf v; } b0a, b0b, b1a, b1b;
        const uint4* p0 = reinterpret_cast<const uint4*>(brow0 + hv * 16);
        const uint4* q0 = reinterpret_cast<const uint4*>(brow0 + 32 + hv * 16);
        const uint4* p1 = reinterpret_cast<const uint4*>(brow1 + hv * 16);
        const uint4* q1 = reinterpret_cast<const uint4*>(brow1 + 32 + hv * 16);
        b0a.u[0] = p0[0]; b0a.u[1] = p0[1];
        b0b.u[0] = q0[0]; b0b.u[1] = q0[1];
        b1a.u[0] = p1[0]; b1a.u[1] = p1[1];
        b1b.u[0] = q1[0]; b1b.u[1] = q1[1];

        const float cq0 = cbsqc[n0 + lrow];
        const float cq1 = cbsqc[n0 + 16 + lrow];

        // Two independent accumulator chains -> interleaved WMMAs fill the
        // bf16 WMMA RAW hazard slots with useful work.
        v8f acc0 = {};
        v8f acc1 = {};
        acc0 = __builtin_amdgcn_wmma_f32_16x16x32_bf16(
                   false, a0, false, b0a.v, (short)0, acc0, false, false);
        acc1 = __builtin_amdgcn_wmma_f32_16x16x32_bf16(
                   false, a0, false, b1a.v, (short)0, acc1, false, false);
        acc0 = __builtin_amdgcn_wmma_f32_16x16x32_bf16(
                   false, a1, false, b0b.v, (short)0, acc0, false, false);
        acc1 = __builtin_amdgcn_wmma_f32_16x16x32_bf16(
                   false, a1, false, b1b.v, (short)0, acc1, false, false);

        // D layout: lane N = lane&15, VGPR r -> row M = r + 8*(lane>=16)
        const int nidx0 = n0 + lrow;
        const int nidx1 = n0 + 16 + lrow;
#pragma unroll
        for (int r = 0; r < 8; ++r) {
            float s0 = fmaf(-2.f, acc0[r], cq0);    // cbsq - 2*dot
            float s1 = fmaf(-2.f, acc1[r], cq1);
            if (s0 < best[r]) { best[r] = s0; bidx[r] = nidx0; }
            if (s1 < best[r]) { best[r] = s1; bidx[r] = nidx1; }
        }
    }

    // ---- Argmin butterfly across each 16-lane N-group ----
#pragma unroll
    for (int r = 0; r < 8; ++r) {
#pragma unroll
        for (int mask = 1; mask < 16; mask <<= 1) {
            float ov = __shfl_xor(best[r], mask, 16);
            int   oi = __shfl_xor(bidx[r], mask, 16);
            if (ov < best[r] || (ov == best[r] && oi < bidx[r])) {
                best[r] = ov; bidx[r] = oi;
            }
        }
    }

    // ---- Gather winning f32 codewords: 16 rows, 64 floats each (2/lane) ----
    const float* cbf = cb + (size_t)c * PQ_N * PQ_SUB;
#pragma unroll
    for (int m = 0; m < 16; ++m) {
        const int srcLane = (m & 8) ? 16 : 0;     // half-wave holding row m
        const int r = m & 7;
        const int idxm = __shfl(bidx[r], srcLane, 32);
        const float2* srcp =
            reinterpret_cast<const float2*>(cbf + (size_t)idxm * PQ_SUB);
        float2* dstp =
            reinterpret_cast<float2*>(out + (size_t)(rowBase + m) * PQ_D + c * PQ_SUB);
        dstp[lane] = srcp[lane];
    }
}

// ---------------------------------------------------------------------------
extern "C" void kernel_launch(void* const* d_in, const int* in_sizes, int n_in,
                              void* d_out, int out_size, void* d_ws, size_t ws_size,
                              hipStream_t stream) {
    (void)in_sizes; (void)n_in; (void)out_size; (void)ws_size;
    const float* x  = (const float*)d_in[0];   // (4,2048,1024) f32
    const float* cb = (const float*)d_in[1];   // (16,2048,64)  f32
    float* out = (float*)d_out;

    __bf16* cb16 = (__bf16*)d_ws;                                   // 4 MiB
    float*  cbsq = (float*)((char*)d_ws +
                            (size_t)PQ_C * PQ_N * PQ_SUB * sizeof(__bf16));

    const int rows = PQ_C * PQ_N;                                   // 32768
    pq_prep_kernel<<<(rows + 255) / 256, 256, 0, stream>>>(cb, cb16, cbsq);

    const int grid = (PQ_M / ROWS_PER_WG) * PQ_C;                   // 1024
    pq_main_kernel<<<grid, 256, 0, stream>>>(x, cb, cb16, cbsq, out);
}
